// GIN_12704513261596
// MI455X (gfx1250) — matrix operand order repaired and checked by reading, not compiled, not run on hardware
//
#include <hip/hip_runtime.h>
#include <hip/hip_bf16.h>

typedef __attribute__((ext_vector_type(16))) __bf16 v16bf;
typedef __attribute__((ext_vector_type(8)))  __bf16 v8bf;
typedef __attribute__((ext_vector_type(8)))  float  v8f;

#define GN 65536
#define GD 128
#define GE 1048576
#define GB 16
#define GNN 4096
#define GCD 512

// ---------------- utility kernels ----------------

__global__ void zero_f32_k(float* __restrict__ p, int n) {
    int i = blockIdx.x * blockDim.x + threadIdx.x;
    if (i < n) p[i] = 0.f;
}

// Pack a K x 128 fp32 row-major weight matrix into WMMA B-fragment order:
// dst[((kt*8 + n)*32 + lane)*16 + e]  where lanes 0-15 hold K=kt*32+e,
// lanes 16-31 hold K=kt*32+16+e, column = n*16 + (lane&15).
__global__ void pack_w_k(const float* __restrict__ src, __bf16* __restrict__ dst, int K) {
    int i = blockIdx.x * blockDim.x + threadIdx.x;
    if (i >= K * 128) return;
    int e    = i & 15;
    int lane = (i >> 4) & 31;
    int fi   = i >> 9;
    int n    = fi & 7;
    int kt   = fi >> 3;
    int krow = kt * 32 + ((lane < 16) ? 0 : 16) + e;
    int col  = n * 16 + (lane & 15);
    dst[i] = (__bf16)src[krow * 128 + col];
}

// h[n][d] = embed[state[n]][d]; also write bf16 copy into cat[:, 0:128]
__global__ void embed_k(const int* __restrict__ state, const float* __restrict__ embed,
                        float* __restrict__ h, __bf16* __restrict__ cat) {
    int i = blockIdx.x * blockDim.x + threadIdx.x;
    if (i >= GN * GD) return;
    int row = i >> 7, c = i & 127;
    float v = embed[state[row] * GD + c];
    h[i] = v;
    cat[(long)row * GCD + c] = (__bf16)v;
}

// scatter-add: 32 threads per edge, float4 per thread
__global__ void scatter_k(const int* __restrict__ src, const int* __restrict__ dst,
                          const float* __restrict__ h, float* __restrict__ agg) {
    long t = (long)blockIdx.x * blockDim.x + threadIdx.x;
    if (t >= (long)GE * 32) return;
    int e = (int)(t >> 5), j = (int)(t & 31);
    int s = src[e], d = dst[e];
    const float4 v = *(const float4*)(h + (long)s * GD + j * 4);
    float* p = agg + (long)d * GD + j * 4;
    atomicAdd(p + 0, v.x);
    atomicAdd(p + 1, v.y);
    atomicAdd(p + 2, v.z);
    atomicAdd(p + 3, v.w);
}

__global__ void add_bf16_k(const float* __restrict__ h, const float* __restrict__ agg,
                           __bf16* __restrict__ zb, int n) {
    int i = blockIdx.x * blockDim.x + threadIdx.x;
    if (i < n) zb[i] = (__bf16)(h[i] + agg[i]);
}

// column sums / sums of squares over an N x 128 matrix (512 rows per block)
__global__ void colstats_k(const float* __restrict__ y, float* __restrict__ cs,
                           float* __restrict__ cq) {
    int c   = threadIdx.x & 127;
    int sub = threadIdx.x >> 7;
    int rb  = blockIdx.x * 512;
    float s = 0.f, q = 0.f;
    for (int r = sub; r < 512; r += 2) {
        float v = y[(long)(rb + r) * GD + c];
        s += v; q += v * v;
    }
    atomicAdd(&cs[c], s);
    atomicAdd(&cq[c], q);
}

__global__ void finalize_stats_k(const float* __restrict__ cs, const float* __restrict__ cq,
                                 const float* __restrict__ g, const float* __restrict__ b,
                                 float* __restrict__ scale, float* __restrict__ shift) {
    int c = threadIdx.x;
    if (c >= GD) return;
    const float invN = 1.f / (float)GN;
    float mean = cs[c] * invN;
    float var  = cq[c] * invN - mean * mean;
    float rs   = rsqrtf(var + 1e-5f);
    float sc   = rs * g[c];
    scale[c] = sc;
    shift[c] = b[c] - mean * sc;
}

// out = relu(y*scale + shift) -> bf16 (strided, for cat or zb2) and optional fp32 h
__global__ void bnrelu_k(const float* __restrict__ y, const float* __restrict__ scale,
                         const float* __restrict__ shift, __bf16* __restrict__ ob,
                         int ostride, int ocol, float* __restrict__ of) {
    int i = blockIdx.x * blockDim.x + threadIdx.x;
    if (i >= GN * GD) return;
    int row = i >> 7, c = i & 127;
    float v = fmaxf(fmaf(y[i], scale[c], shift[c]), 0.f);
    ob[(long)row * ostride + ocol + c] = (__bf16)v;
    if (of) of[i] = v;
}

// ---------------- WMMA GEMM ----------------
// Block = 256 threads = 8 waves; each wave computes a 16-row x 128-col strip.
// A: bf16 row-major (lda elements per row), W: prepacked fragments, K in {128,512}.
// MODE 0: write f32 N x 128.  MODE 1: fused readout -> score[row].
template <int MODE>
__global__ void gemm_bf16_k(const __bf16* __restrict__ A, int lda,
                            const __bf16* __restrict__ Wp, int K,
                            float* __restrict__ out,
                            const float* __restrict__ br1,
                            const float* __restrict__ wr2,
                            const float* __restrict__ br2) {
    const int lane = threadIdx.x & 31;
    const int wv   = threadIdx.x >> 5;
    const int m0   = blockIdx.x * 128 + wv * 16;

    v8f acc[8];
#pragma unroll
    for (int n = 0; n < 8; ++n)
#pragma unroll
        for (int j = 0; j < 8; ++j) acc[n][j] = 0.f;

    const int row = m0 + (lane & 15);
    const int c0  = (lane < 16) ? 0 : 16;
    const __bf16* arow = A + (long)row * lda;
    const int nkt = K >> 5;

    union AV { v16bf v; v8bf h[2]; };

    for (int kt = 0; kt < nkt; ++kt) {
        AV a;
        a.h[0] = *(const v8bf*)(arow + kt * 32 + c0);
        a.h[1] = *(const v8bf*)(arow + kt * 32 + c0 + 8);
        const __bf16* wb = Wp + ((long)(kt * 8) * 32 + lane) * 16;
#pragma unroll
        for (int n = 0; n < 8; ++n) {
            v16bf b = *(const v16bf*)(wb + n * 512);
            acc[n] = __builtin_amdgcn_wmma_f32_16x16x32_bf16(
                false, a.v, false, b, (short)0, acc[n], false, false);
        }
    }

    if (MODE == 0) {
        const int rbase = m0 + ((lane < 16) ? 0 : 8);
        const int cl = lane & 15;
#pragma unroll
        for (int n = 0; n < 8; ++n)
#pragma unroll
            for (int j = 0; j < 8; ++j)
                out[(long)(rbase + j) * GD + n * 16 + cl] = acc[n][j];
    } else {
        // score = relu(acc + br1) . wr2 + br2, one scalar per row
        const int cl = lane & 15;
        float part[8];
#pragma unroll
        for (int j = 0; j < 8; ++j) part[j] = 0.f;
#pragma unroll
        for (int n = 0; n < 8; ++n) {
            int c = n * 16 + cl;
            float bb = br1[c], sw = wr2[c];
#pragma unroll
            for (int j = 0; j < 8; ++j)
                part[j] += fmaxf(acc[n][j] + bb, 0.f) * sw;
        }
        float b2 = br2[0];
#pragma unroll
        for (int j = 0; j < 8; ++j) {
            float p = part[j];
            p += __shfl_xor(p, 1, 32);
            p += __shfl_xor(p, 2, 32);
            p += __shfl_xor(p, 4, 32);
            p += __shfl_xor(p, 8, 32);
            if (cl == 0) {
                int r = m0 + j + ((lane < 16) ? 0 : 8);
                out[r] = p + b2;
            }
        }
    }
}

// per-graph max over 4096 scores
__global__ void segmax_k(const float* __restrict__ score, float* __restrict__ pooled) {
    __shared__ float red[256];
    int g = blockIdx.x;
    float m = -3.402823466e+38f;
    for (int i = threadIdx.x; i < GNN; i += 256)
        m = fmaxf(m, score[(long)g * GNN + i]);
    red[threadIdx.x] = m;
    __syncthreads();
    for (int s = 128; s > 0; s >>= 1) {
        if (threadIdx.x < s) red[threadIdx.x] = fmaxf(red[threadIdx.x], red[threadIdx.x + s]);
        __syncthreads();
    }
    if (threadIdx.x == 0) pooled[g] = red[0];
}

// ---------------- launcher ----------------

extern "C" void kernel_launch(void* const* d_in, const int* in_sizes, int n_in,
                              void* d_out, int out_size, void* d_ws, size_t ws_size,
                              hipStream_t stream) {
    const int N = GN, D = GD, E = GE, B = GB, CD = GCD;

    const int*   state = (const int*)d_in[0];
    const int*   ei    = (const int*)d_in[1];
    const float* emb   = (const float*)d_in[2];

    const float *w1[3], *w2[3], *g1[3], *b1[3], *g2[3], *b2[3];
    const float *wr1, *br1, *wr2, *br2;
    if (n_in >= 25) {
        for (int i = 0; i < 3; ++i) {
            w1[i] = (const float*)d_in[3 + i];
            w2[i] = (const float*)d_in[6 + i];
            g1[i] = (const float*)d_in[9 + i];
            b1[i] = (const float*)d_in[12 + i];
            g2[i] = (const float*)d_in[15 + i];
            b2[i] = (const float*)d_in[18 + i];
        }
        wr1 = (const float*)d_in[21];
        br1 = (const float*)d_in[22];
        wr2 = (const float*)d_in[23];
        br2 = (const float*)d_in[24];
    } else {
        // lists stacked as single arrays
        const float* W1 = (const float*)d_in[3];
        const float* W2 = (const float*)d_in[4];
        const float* G1 = (const float*)d_in[5];
        const float* B1 = (const float*)d_in[6];
        const float* G2 = (const float*)d_in[7];
        const float* B2 = (const float*)d_in[8];
        for (int i = 0; i < 3; ++i) {
            w1[i] = W1 + i * D * D;
            w2[i] = W2 + i * D * D;
            g1[i] = G1 + i * D;
            b1[i] = B1 + i * D;
            g2[i] = G2 + i * D;
            b2[i] = B2 + i * D;
        }
        wr1 = (const float*)d_in[9];
        br1 = (const float*)d_in[10];
        wr2 = (const float*)d_in[11];
        br2 = (const float*)d_in[12];
    }

    // workspace carve-up (all offsets 256B aligned)
    char* ws = (char*)d_ws;
    float*  h      = (float*)(ws + 0);                 // N*D*4   = 33.55 MB
    float*  ybuf   = (float*)(ws + 33554432);          // N*D*4   (agg / y1 / y2 reuse)
    __bf16* zb     = (__bf16*)(ws + 67108864);         // N*D*2
    __bf16* zb2    = (__bf16*)(ws + 83886080);         // N*D*2
    __bf16* cat    = (__bf16*)(ws + 100663296);        // N*512*2 = 67.1 MB
    __bf16* wpack  = (__bf16*)(ws + 167772160);        // 6 * D*D bf16
    __bf16* wrpack = (__bf16*)(ws + 167968768);        // CD*D bf16
    float*  colsum = (float*)(ws + 168099840);
    float*  colsq  = colsum + 128;
    float*  scale  = colsum + 256;
    float*  shift  = colsum + 384;

    const int T = 256;

    // 1) pack weights to WMMA fragment layout (bf16)
    for (int i = 0; i < 3; ++i) {
        pack_w_k<<<(D * D + T - 1) / T, T, 0, stream>>>(w1[i], wpack + (2 * i) * D * D, D);
        pack_w_k<<<(D * D + T - 1) / T, T, 0, stream>>>(w2[i], wpack + (2 * i + 1) * D * D, D);
    }
    pack_w_k<<<(CD * D + T - 1) / T, T, 0, stream>>>(wr1, wrpack, CD);

    // 2) embedding lookup -> h, cat[:,0:128]
    embed_k<<<(N * D + T - 1) / T, T, 0, stream>>>(state, emb, h, cat);

    // 3) GIN layers
    for (int i = 0; i < 3; ++i) {
        zero_f32_k<<<(N * D + T - 1) / T, T, 0, stream>>>(ybuf, N * D);
        scatter_k<<<(int)(((long)E * 32 + T - 1) / T), T, 0, stream>>>(ei, ei + E, h, ybuf);
        add_bf16_k<<<(N * D + T - 1) / T, T, 0, stream>>>(h, ybuf, zb, N * D);

        // z @ w1 -> ybuf
        gemm_bf16_k<0><<<N / 128, T, 0, stream>>>(zb, D, wpack + (2 * i) * D * D, D,
                                                  ybuf, nullptr, nullptr, nullptr);
        zero_f32_k<<<1, T, 0, stream>>>(colsum, 256);
        colstats_k<<<N / 512, T, 0, stream>>>(ybuf, colsum, colsq);
        finalize_stats_k<<<1, 128, 0, stream>>>(colsum, colsq, g1[i], b1[i], scale, shift);
        bnrelu_k<<<(N * D + T - 1) / T, T, 0, stream>>>(ybuf, scale, shift, zb2, D, 0, nullptr);

        // relu(bn(..)) @ w2 -> ybuf
        gemm_bf16_k<0><<<N / 128, T, 0, stream>>>(zb2, D, wpack + (2 * i + 1) * D * D, D,
                                                  ybuf, nullptr, nullptr, nullptr);
        zero_f32_k<<<1, T, 0, stream>>>(colsum, 256);
        colstats_k<<<N / 512, T, 0, stream>>>(ybuf, colsum, colsq);
        finalize_stats_k<<<1, 128, 0, stream>>>(colsum, colsq, g2[i], b2[i], scale, shift);
        // h = relu(bn2(y2)) fp32, plus bf16 copy into cat slice (i+1)
        bnrelu_k<<<(N * D + T - 1) / T, T, 0, stream>>>(ybuf, scale, shift,
                                                        cat, CD, (i + 1) * D, h);
    }

    // 4) readout: score = relu(cat @ wr1 + br1) @ wr2 + br2  (fused epilogue)
    float* score = (float*)d_out;
    gemm_bf16_k<1><<<N / 128, T, 0, stream>>>(cat, CD, wrpack, CD, score, br1, wr2, br2);

    // 5) per-graph segment max -> d_out[N .. N+B)
    segmax_k<<<B, T, 0, stream>>>(score, score + N);
}
